// AttentionHead_63660005261375
// MI455X (gfx1250) — matrix operand order repaired and checked by reading, not compiled, not run on hardware
//
#include <hip/hip_runtime.h>

// Problem constants (match reference): B=4, S=4096, H=2048, D=128
constexpr int BB = 4;
constexpr int SS = 4096;
constexpr int HH = 2048;
constexpr int DD = 128;

typedef __attribute__((ext_vector_type(16))) __bf16 v16bf;
typedef __attribute__((ext_vector_type(8)))  float  v8f;
typedef __attribute__((ext_vector_type(4)))  float  v4f;

union Frag16 {
  v16bf v;
  v4f   f4[2];
};

// fp32 -> bf16 via native convert (RNE on CDNA5, lowers to v_cvt_pk_bf16_f32)
__device__ __forceinline__ unsigned short f2bf(float f) {
  __bf16 h = (__bf16)f;
  return __builtin_bit_cast(unsigned short, h);
}

// LDS row pitch in bf16 elements: 40*2 = 80 bytes (16B-aligned rows, banks spread)
constexpr int LP = 40;

// Low 32 bits of a flat shared-memory address are the LDS byte offset
// (ISA 10.2: LDS aperture maps addr[31:0] -> LDS offset).
__device__ __forceinline__ unsigned lds_off(const void* p) {
  return (unsigned)(unsigned long long)p;
}

// Async global->LDS 16-byte copy (CDNA5, ASYNCcnt-tracked).
__device__ __forceinline__ void async_cp16(unsigned lds, const void* g) {
  asm volatile("global_load_async_to_lds_b128 %0, %1, off"
               :: "v"(lds), "v"((unsigned long long)g)
               : "memory");
}

// ---------------------------------------------------------------------------
// Kernel 1: q/k/v projections.  C(16384x128) = X(16384x2048) @ W(2048x128)
// blockIdx.x: 128-row strip, blockIdx.z: which weight matrix (0=q,1=k,2=v)
// ---------------------------------------------------------------------------
__global__ __launch_bounds__(256) void proj_kernel(
    const float* __restrict__ x,
    const float* __restrict__ Wq,
    const float* __restrict__ Wk,
    const float* __restrict__ Wv,
    unsigned short* __restrict__ qb,
    unsigned short* __restrict__ kb,
    unsigned short* __restrict__ vb)
{
  __shared__ alignas(16) unsigned short As[128 * LP];   // x tile   [m=128][k=32]
  __shared__ alignas(16) unsigned short Bt[128 * LP];   // W tile^T [n=128][k=32]

  const int tid  = threadIdx.x;
  const int lane = tid & 31;
  const int wv   = tid >> 5;      // wave 0..7
  const int ml   = lane & 15;     // row/col within 16
  const int hs   = lane >> 4;     // half-select
  const int row0 = blockIdx.x * 128;

  const float* Wm;
  unsigned short* ob;
  if (blockIdx.z == 0)      { Wm = Wq; ob = qb; }
  else if (blockIdx.z == 1) { Wm = Wk; ob = kb; }
  else                      { Wm = Wv; ob = vb; }

  v8f acc[8];
  #pragma unroll
  for (int j = 0; j < 8; ++j)
    #pragma unroll
    for (int i = 0; i < 8; ++i) acc[j][i] = 0.0f;

  for (int k0 = 0; k0 < HH; k0 += 32) {
    #pragma unroll
    for (int it = 0; it < 16; ++it) {
      int r = (tid >> 5) + it * 8;
      int c = tid & 31;
      As[r * LP + c] = f2bf(x[(size_t)(row0 + r) * HH + (k0 + c)]);
    }
    #pragma unroll
    for (int it = 0; it < 16; ++it) {
      int k = (tid >> 7) + it * 2;
      int n = tid & 127;
      Bt[n * LP + k] = f2bf(Wm[(size_t)(k0 + k) * DD + n]);
    }
    __syncthreads();

    Frag16 af;
    const unsigned short* ar = &As[(wv * 16 + ml) * LP];
    af.f4[0] = *(const v4f*)(ar + 16 * hs);
    af.f4[1] = *(const v4f*)(ar + 16 * hs + 8);

    // Double-buffered B fragments: overlap ds_load of frag j+1 with WMMA j.
    Frag16 bb[2];
    {
      const unsigned short* br = &Bt[ml * LP + 16 * hs];
      bb[0].f4[0] = *(const v4f*)(br);
      bb[0].f4[1] = *(const v4f*)(br + 8);
    }
    #pragma unroll
    for (int j = 0; j < 8; ++j) {
      if (j < 7) {
        const unsigned short* br = &Bt[((j + 1) * 16 + ml) * LP + 16 * hs];
        bb[(j + 1) & 1].f4[0] = *(const v4f*)(br);
        bb[(j + 1) & 1].f4[1] = *(const v4f*)(br + 8);
      }
      acc[j] = __builtin_amdgcn_wmma_f32_16x16x32_bf16(
          false, af.v, false, bb[j & 1].v, (short)0, acc[j], false, false);
    }
    __syncthreads();
  }

  #pragma unroll
  for (int j = 0; j < 8; ++j) {
    int n = j * 16 + ml;
    #pragma unroll
    for (int r = 0; r < 8; ++r) {
      int m = wv * 16 + r + 8 * hs;
      ob[(size_t)(row0 + m) * DD + n] = f2bf(acc[j][r]);
    }
  }
}

// ---------------------------------------------------------------------------
// Kernel 2: masked scores.  S(b) = (Q(b) @ K(b)^T) * scale, mask -> -inf
// Ping-pong LDS buffers with async global->LDS DMA overlapped with WMMA.
// ---------------------------------------------------------------------------
__global__ __launch_bounds__(256) void scores_kernel(
    const unsigned short* __restrict__ qb,
    const unsigned short* __restrict__ kb,
    const int* __restrict__ msk,
    float* __restrict__ wout)
{
  __shared__ alignas(16) unsigned short As[2][128 * LP];
  __shared__ alignas(16) unsigned short Bt[2][128 * LP];

  const int tid  = threadIdx.x;
  const int lane = tid & 31;
  const int wv   = tid >> 5;
  const int ml   = lane & 15;
  const int hs   = lane >> 4;
  const int b    = blockIdx.z;
  const int q0   = blockIdx.y * 128;
  const int n0   = blockIdx.x * 128;

  const unsigned short* Qp = qb + (size_t)b * SS * DD;
  const unsigned short* Kp = kb + (size_t)b * SS * DD;

  const unsigned asB[2] = { lds_off(&As[0][0]), lds_off(&As[1][0]) };
  const unsigned btB[2] = { lds_off(&Bt[0][0]), lds_off(&Bt[1][0]) };

  // Issue the 4 async 16B copies this thread owns for one (As,Bt) tile pair.
  auto issue = [&](int buf, int k0) {
    #pragma unroll
    for (int t = 0; t < 2; ++t) {
      int id = tid + t * 256;
      int r = id >> 2;        // row 0..127
      int c = id & 3;         // 16B chunk 0..3
      async_cp16(asB[buf] + (unsigned)(r * (LP * 2) + c * 16),
                 Qp + (size_t)(q0 + r) * DD + k0 + c * 8);
      async_cp16(btB[buf] + (unsigned)(r * (LP * 2) + c * 16),
                 Kp + (size_t)(n0 + r) * DD + k0 + c * 8);
    }
  };

  v8f acc[8];
  #pragma unroll
  for (int j = 0; j < 8; ++j)
    #pragma unroll
    for (int i = 0; i < 8; ++i) acc[j][i] = 0.0f;

  issue(0, 0);
  #pragma unroll
  for (int ki = 0; ki < 4; ++ki) {      // D/32 = 4 K-steps
    const int cur = ki & 1;
    if (ki < 3) {
      issue(cur ^ 1, (ki + 1) * 32);
      // 4 ops just issued are outstanding; <=4 means previous tile's 4
      // ops retired (async loads complete in order, ISA 08 §4.1).
      asm volatile("s_wait_asynccnt 0x4" ::: "memory");
    } else {
      asm volatile("s_wait_asynccnt 0x0" ::: "memory");
    }
    __syncthreads();

    Frag16 af;
    const unsigned short* ar = &As[cur][(wv * 16 + ml) * LP];
    af.f4[0] = *(const v4f*)(ar + 16 * hs);
    af.f4[1] = *(const v4f*)(ar + 16 * hs + 8);

    Frag16 bb[2];
    {
      const unsigned short* br = &Bt[cur][ml * LP + 16 * hs];
      bb[0].f4[0] = *(const v4f*)(br);
      bb[0].f4[1] = *(const v4f*)(br + 8);
    }
    #pragma unroll
    for (int j = 0; j < 8; ++j) {
      if (j < 7) {
        const unsigned short* br = &Bt[cur][((j + 1) * 16 + ml) * LP + 16 * hs];
        bb[(j + 1) & 1].f4[0] = *(const v4f*)(br);
        bb[(j + 1) & 1].f4[1] = *(const v4f*)(br + 8);
      }
      acc[j] = __builtin_amdgcn_wmma_f32_16x16x32_bf16(
          false, af.v, false, bb[j & 1].v, (short)0, acc[j], false, false);
    }
    __syncthreads();
  }

  const float scale = 0.08838834764831845f;  // 1/sqrt(128)
  #pragma unroll
  for (int j = 0; j < 8; ++j) {
    int n = n0 + j * 16 + ml;
    #pragma unroll
    for (int r = 0; r < 8; ++r) {
      int m = q0 + wv * 16 + r + 8 * hs;
      size_t idx = ((size_t)b * SS + m) * SS + n;
      float sv = acc[j][r] * scale;
      wout[idx] = msk[idx] ? sv : -__builtin_inff();
    }
  }
}

// ---------------------------------------------------------------------------
// Kernel 3: per-row stats (max, 1/sum of exp). 1 block per row; read-only.
// ---------------------------------------------------------------------------
__global__ __launch_bounds__(256) void rowstats_kernel(
    const float* __restrict__ wts,
    float* __restrict__ mxA,
    float* __restrict__ invA)
{
  __shared__ float red[256];
  const int tid = threadIdx.x;
  const float* p = wts + (size_t)blockIdx.x * SS;

  float mx = -__builtin_inff();
  for (int i = tid; i < SS; i += 256) mx = fmaxf(mx, p[i]);
  red[tid] = mx;
  __syncthreads();
  for (int s = 128; s > 0; s >>= 1) {
    if (tid < s) red[tid] = fmaxf(red[tid], red[tid + s]);
    __syncthreads();
  }
  mx = red[0];
  __syncthreads();

  float sum = 0.0f;
  for (int i = tid; i < SS; i += 256) sum += __expf(p[i] - mx);
  red[tid] = sum;
  __syncthreads();
  for (int s = 128; s > 0; s >>= 1) {
    if (tid < s) red[tid] += red[tid + s];
    __syncthreads();
  }
  if (tid == 0) {
    mxA[blockIdx.x]  = mx;
    invA[blockIdx.x] = 1.0f / red[0];
  }
}

// ---------------------------------------------------------------------------
// Kernel 4: normalize weights in place AND compute output = P @ V.
// Each weights element is staged exactly once -> write normalized fp32 back
// to d_out and the bf16 copy to LDS in the same pass.
// ---------------------------------------------------------------------------
__global__ __launch_bounds__(256) void av_kernel(
    float* wts,                                   // raw in, normalized out
    const unsigned short* __restrict__ vb,
    const float* __restrict__ mxA,
    const float* __restrict__ invA,
    float* __restrict__ out)
{
  __shared__ alignas(16) unsigned short As[128 * LP];
  __shared__ alignas(16) unsigned short Bt[128 * LP];

  const int tid  = threadIdx.x;
  const int lane = tid & 31;
  const int wv   = tid >> 5;
  const int ml   = lane & 15;
  const int hs   = lane >> 4;
  const int b    = blockIdx.z;
  const int q0   = blockIdx.x * 128;

  float* Pp = wts + (size_t)b * SS * SS;
  const unsigned short* Vp = vb + (size_t)b * SS * DD;

  // Preload the row stats this thread needs (rows r = tid>>5 + it*8).
  float mxv[16], inv[16];
  #pragma unroll
  for (int it = 0; it < 16; ++it) {
    int r = (tid >> 5) + it * 8;
    size_t row = (size_t)b * SS + q0 + r;
    mxv[it] = mxA[row];
    inv[it] = invA[row];
  }

  v8f acc[8];
  #pragma unroll
  for (int j = 0; j < 8; ++j)
    #pragma unroll
    for (int i = 0; i < 8; ++i) acc[j][i] = 0.0f;

  for (int k0 = 0; k0 < SS; k0 += 32) {
    #pragma unroll
    for (int it = 0; it < 16; ++it) {
      int r = (tid >> 5) + it * 8;
      int c = tid & 31;
      size_t gidx = (size_t)(q0 + r) * SS + (k0 + c);
      float e = __expf(Pp[gidx] - mxv[it]) * inv[it];
      Pp[gidx] = e;
      As[r * LP + c] = f2bf(e);
    }
    #pragma unroll
    for (int it = 0; it < 16; ++it) {
      int k = (tid >> 7) + it * 2;
      int n = tid & 127;
      Bt[n * LP + k] = Vp[(size_t)(k0 + k) * DD + n];
    }
    __syncthreads();

    Frag16 af;
    const unsigned short* ar = &As[(wv * 16 + ml) * LP];
    af.f4[0] = *(const v4f*)(ar + 16 * hs);
    af.f4[1] = *(const v4f*)(ar + 16 * hs + 8);

    Frag16 bb[2];
    {
      const unsigned short* br = &Bt[ml * LP + 16 * hs];
      bb[0].f4[0] = *(const v4f*)(br);
      bb[0].f4[1] = *(const v4f*)(br + 8);
    }
    #pragma unroll
    for (int j = 0; j < 8; ++j) {
      if (j < 7) {
        const unsigned short* br = &Bt[((j + 1) * 16 + ml) * LP + 16 * hs];
        bb[(j + 1) & 1].f4[0] = *(const v4f*)(br);
        bb[(j + 1) & 1].f4[1] = *(const v4f*)(br + 8);
      }
      acc[j] = __builtin_amdgcn_wmma_f32_16x16x32_bf16(
          false, af.v, false, bb[j & 1].v, (short)0, acc[j], false, false);
    }
    __syncthreads();
  }

  #pragma unroll
  for (int j = 0; j < 8; ++j) {
    int n = j * 16 + ml;
    #pragma unroll
    for (int r = 0; r < 8; ++r) {
      int m = q0 + wv * 16 + r + 8 * hs;
      out[((size_t)b * SS + m) * DD + n] = acc[j][r];
    }
  }
}

// ---------------------------------------------------------------------------
extern "C" void kernel_launch(void* const* d_in, const int* in_sizes, int n_in,
                              void* d_out, int out_size, void* d_ws, size_t ws_size,
                              hipStream_t stream)
{
  const float* x   = (const float*)d_in[0];
  const float* Wq  = (const float*)d_in[1];
  const float* Wk  = (const float*)d_in[2];
  const float* Wv  = (const float*)d_in[3];
  const int*   msk = (const int*)d_in[4];

  float* out = (float*)d_out;                       // (B,S,D) fp32
  float* wts = out + (size_t)BB * SS * DD;          // (B,S,S) fp32

  // workspace: q/k/v bf16 (12 MB) + row stats (128 KB)
  unsigned short* qb = (unsigned short*)d_ws;
  unsigned short* kb = qb + (size_t)BB * SS * DD;
  unsigned short* vb = kb + (size_t)BB * SS * DD;
  float* mxA  = (float*)(vb + (size_t)BB * SS * DD);
  float* invA = mxA + (size_t)BB * SS;

  proj_kernel<<<dim3((BB * SS) / 128, 1, 3), 256, 0, stream>>>(
      x, Wq, Wk, Wv, qb, kb, vb);
  scores_kernel<<<dim3(SS / 128, SS / 128, BB), 256, 0, stream>>>(
      qb, kb, msk, wts);
  rowstats_kernel<<<dim3(BB * SS), 256, 0, stream>>>(wts, mxA, invA);
  av_kernel<<<dim3(SS / 128, 1, BB), 256, 0, stream>>>(
      wts, vb, mxA, invA, out);
}
